// Attention_44933947850837
// MI455X (gfx1250) — matrix operand order repaired and checked by reading, not compiled, not run on hardware
//
#include <hip/hip_runtime.h>
#include <hip/hip_bf16.h>

// ---------------------------------------------------------------------------
// Types for WMMA fragments / TDM descriptors (trivial ext_vector types only)
// ---------------------------------------------------------------------------
typedef __attribute__((ext_vector_type(16))) __bf16 v16bf;
typedef __attribute__((ext_vector_type(8)))  float  v8f;
typedef __attribute__((ext_vector_type(4)))  unsigned int u32x4;
typedef __attribute__((ext_vector_type(8)))  int    i32x8;
typedef __attribute__((ext_vector_type(4)))  int    i32x4;

union AFrag { v16bf v; u32x4 q[2]; };

static __device__ __forceinline__ __bf16 f2bf(float f) {
  unsigned u = __builtin_bit_cast(unsigned, f);
  unsigned r = (u + 0x7FFFu + ((u >> 16) & 1u)) >> 16;   // round-to-nearest-even
  return __builtin_bit_cast(__bf16, (unsigned short)r);
}

static __device__ __forceinline__ v8f wmma_bf16(const AFrag& a, const AFrag& b, v8f c) {
  return __builtin_amdgcn_wmma_f32_16x16x32_bf16(
      /*neg_a=*/false, a.v, /*neg_b=*/false, b.v,
      /*c_mod=*/(short)0, c, /*reuse_a=*/false, /*reuse_b=*/false);
}

// Problem constants
#define BATCH 8
#define SEQ   1024
#define DMODEL 768
#define NHEAD 12
#define HDIM  64
#define ROWS  (BATCH * SEQ)     // 8192

// ---------------------------------------------------------------------------
// Tensor Data Mover: 2D bf16 tile load Global -> LDS (6-arg builtin on this
// toolchain: g0, g1, g2, g3, ext-group, cpol). Descriptor per CDNA5 ISA ch.8:
// group0 = {count/lds_addr/global_addr/type}, group1 = {data_size, tensor
// dims, tile dims, dim0 stride}. Groups 2/3/ext zero (2D tensor, tile_dim2=0).
// ---------------------------------------------------------------------------
#if defined(__has_builtin)
#if __has_builtin(__builtin_amdgcn_tensor_load_to_lds) && \
    __has_builtin(__builtin_amdgcn_s_wait_tensorcnt)
#define USE_TDM 1
#endif
#endif
#ifndef USE_TDM
#define USE_TDM 0
#endif

#if USE_TDM
static __device__ __forceinline__ void tdm_load_2d_bf16(
    unsigned lds_addr, const void* gptr,
    unsigned tensor_d0, unsigned tensor_d1,
    unsigned tile_d0, unsigned tile_d1,
    unsigned stride0_elems) {
  unsigned long long ga = (unsigned long long)gptr;
  u32x4 g0;
  g0[0] = 1u;                                              // count=1, user D#
  g0[1] = lds_addr;                                        // LDS byte address
  g0[2] = (unsigned)(ga & 0xffffffffu);                    // global addr lo
  g0[3] = (unsigned)((ga >> 32) & 0x01ffffffu) | 0x80000000u; // addr hi | type=2
  i32x8 g1;
  g1[0] = (int)(1u << 16);                                 // data_size=1 (2B)
  g1[1] = (int)(tensor_d0 << 16);                          // tensor_dim0 lo16
  g1[2] = (int)((tensor_d0 >> 16) | (tensor_d1 << 16));    // dim0 hi | dim1 lo
  g1[3] = (int)((tensor_d1 >> 16) | (tile_d0 << 16));      // dim1 hi | tile0
  g1[4] = (int)(tile_d1 & 0xffffu);                        // tile1 | tile2=0
  g1[5] = (int)stride0_elems;                              // dim0 stride lo32
  g1[6] = 0;                                               // stride hi bits
  g1[7] = 0;
  i32x4 z4 = {0, 0, 0, 0};
  i32x8 z8 = {0, 0, 0, 0, 0, 0, 0, 0};
  __builtin_amdgcn_tensor_load_to_lds(g0, g1, z4, z4, z8, 0);
}
#endif

// ---------------------------------------------------------------------------
// Stage 0a: f32 -> bf16 flat convert (for x)
// ---------------------------------------------------------------------------
__global__ void cvt_f32_bf16_k(const float* __restrict__ in, __bf16* __restrict__ out, int n) {
  int i = blockIdx.x * blockDim.x + threadIdx.x;
  if (i < n) out[i] = f2bf(in[i]);
}

// ---------------------------------------------------------------------------
// Stage 0b: pre-swizzle weight [K, Ncol] (row-major f32) into B-fragment tiles:
//   out[(((kt*NT + nt)*32 + lane)*16 + i)] = W[kt*32 + (i>>3)*16 + (lane>>4)*8 + (i&7)][nt*16 + (lane&15)]
// ---------------------------------------------------------------------------
__global__ void swizzle_w_k(const float* __restrict__ W, __bf16* __restrict__ out,
                            int K, int Ncol) {
  int idx = blockIdx.x * blockDim.x + threadIdx.x;
  if (idx >= K * Ncol) return;
  int i    = idx & 15;
  int lane = (idx >> 4) & 31;
  int tile = idx >> 9;
  int NT   = Ncol >> 4;
  int nt   = tile % NT;
  int kt   = tile / NT;
  int krow = kt * 32 + ((i >> 3) * 16) + ((lane >> 4) * 8) + (i & 7);
  int col  = nt * 16 + (lane & 15);
  out[idx] = f2bf(W[(size_t)krow * Ncol + col]);
}

// ---------------------------------------------------------------------------
// Stage 1: QKV GEMM [8192x768]x[768x2304]+bias, fused scatter to
//   Q [B,H,N,64] (pre-scaled 1/8), K [B,H,N,64], Vt [B,H,64,N]  (bf16).
// Register double-buffered over the K dimension to hide VMEM latency.
// ---------------------------------------------------------------------------
__global__ __launch_bounds__(256) void qkv_gemm_k(
    const __bf16* __restrict__ xb, const __bf16* __restrict__ wsw,
    const float* __restrict__ bias,
    __bf16* __restrict__ Qb, __bf16* __restrict__ Kb, __bf16* __restrict__ Vtb) {
  const int NT = 2304 / 16;                 // 144 column tiles in swizzled W
  int mblk = blockIdx.x;                    // 64
  int nblk = blockIdx.y;                    // 36
  int w    = threadIdx.x >> 5;
  int lane = threadIdx.x & 31;
  int kb8  = (lane >> 4) * 8;
  const __bf16* arow_p = xb + (size_t)(mblk * 128 + w * 16 + (lane & 15)) * DMODEL + kb8;
  const __bf16* bbase  = wsw + ((size_t)(nblk * 4) * 32 + lane) * 16;

  auto loadA = [&](int ks, AFrag& a) {
    const __bf16* ap = arow_p + ks * 32;
    a.q[0] = *(const u32x4*)ap;
    a.q[1] = *(const u32x4*)(ap + 16);
  };
  auto loadB = [&](int ks, int j, AFrag& b) {
    const __bf16* bp = bbase + ((size_t)ks * NT + j) * 512;
    b.q[0] = *(const u32x4*)bp;
    b.q[1] = *(const u32x4*)(bp + 8);
  };

  v8f acc[4] = {};
  AFrag a0, a1, b0[4], b1[4];
  loadA(0, a0);
#pragma unroll
  for (int j = 0; j < 4; ++j) loadB(0, j, b0[j]);

  for (int ks = 0; ks < DMODEL / 32; ks += 2) {
    __builtin_prefetch(arow_p + (ks + 4) * 32, 0, 0);
    __builtin_prefetch(bbase + ((size_t)(ks + 4) * NT) * 512, 0, 0);
    loadA(ks + 1, a1);
#pragma unroll
    for (int j = 0; j < 4; ++j) loadB(ks + 1, j, b1[j]);
#pragma unroll
    for (int j = 0; j < 4; ++j) acc[j] = wmma_bf16(a0, b0[j], acc[j]);
    if (ks + 2 < DMODEL / 32) {
      loadA(ks + 2, a0);
#pragma unroll
      for (int j = 0; j < 4; ++j) loadB(ks + 2, j, b0[j]);
    }
#pragma unroll
    for (int j = 0; j < 4; ++j) acc[j] = wmma_bf16(a1, b1[j], acc[j]);
  }

  int rbase = mblk * 128 + w * 16 + 8 * (lane >> 4);
#pragma unroll
  for (int j = 0; j < 4; ++j) {
    int col = nblk * 64 + j * 16 + (lane & 15);
    float bv = bias[col];
#pragma unroll
    for (int r = 0; r < 8; ++r) {
      int rw = rbase + r;
      int bi = rw >> 10;            // batch
      int n  = rw & 1023;           // token
      float v = acc[j][r] + bv;
      if (col < 768) {              // Q (pre-scale by HD^-0.5 = 0.125)
        int h = col >> 6, hd = col & 63;
        Qb[((((size_t)bi * NHEAD + h) << 10) + n) * HDIM + hd] = f2bf(v * 0.125f);
      } else if (col < 1536) {      // K
        int c = col - 768; int h = c >> 6, hd = c & 63;
        Kb[((((size_t)bi * NHEAD + h) << 10) + n) * HDIM + hd] = f2bf(v);
      } else {                      // V, stored transposed [hd][n]
        int c = col - 1536; int h = c >> 6, hd = c & 63;
        Vtb[((((size_t)bi * NHEAD + h) * HDIM + hd) << 10) + n] = f2bf(v);
      }
    }
  }
}

// ---------------------------------------------------------------------------
// Stage 2: flash attention per (b,h). 8 waves own 128 query rows; 32-key tiles
// double-buffered in LDS, staged by the Tensor Data Mover (wave 0 issues the
// DMA for tile kt+1 while all waves compute tile kt; s_wait_tensorcnt +
// barrier for consumption). Online softmax in f32 registers.
// ---------------------------------------------------------------------------
__global__ __launch_bounds__(256) void attn_k(
    const __bf16* __restrict__ Qb, const __bf16* __restrict__ Kb,
    const __bf16* __restrict__ Vtb, __bf16* __restrict__ Ob) {
  __shared__ __align__(16) __bf16 kT[2][32 * 64];   // keys x hd (double buffer)
  __shared__ __align__(16) __bf16 vT[2][64 * 32];   // hd x keys (double buffer)
  __shared__ __align__(16) __bf16 pT[8][16 * 32];   // per-wave P scratch

  int bh   = blockIdx.x;                  // 96 = B*H
  int qblk = blockIdx.y;                  // 8
  const __bf16* Qh = Qb  + (size_t)bh * SEQ * HDIM;
  const __bf16* Kh = Kb  + (size_t)bh * SEQ * HDIM;
  const __bf16* Vh = Vtb + (size_t)bh * HDIM * SEQ;

  int w    = threadIdx.x >> 5;
  int lane = threadIdx.x & 31;
  int qrow = qblk * 128 + w * 16 + (lane & 15);
  int kb8  = (lane >> 4) * 8;
  int kb16 = (lane >> 4) * 8;

  // Q A-fragments for hd 0..31 and 32..63, resident across the key loop.
  AFrag aq0, aq1;
  {
    const __bf16* p = Qh + (size_t)qrow * HDIM + kb8;
    aq0.q[0] = *(const u32x4*)p;        aq0.q[1] = *(const u32x4*)(p + 16);
    aq1.q[0] = *(const u32x4*)(p + 32); aq1.q[1] = *(const u32x4*)(p + 48);
  }

  float m[8], l[8];
  v8f o[4] = {};
#pragma unroll
  for (int r = 0; r < 8; ++r) { m[r] = -1e30f; l[r] = 0.f; }

  const int NKT = SEQ / 32;
#if USE_TDM
  if (w == 0) {   // prologue: DMA tile 0 into buffer 0
    tdm_load_2d_bf16((unsigned)(unsigned long long)(const void*)&kT[0][0],
                     Kh, 64, 32, 64, 32, 64);
    tdm_load_2d_bf16((unsigned)(unsigned long long)(const void*)&vT[0][0],
                     Vh, 32, 64, 32, 64, SEQ);
  }
#endif

  for (int kt = 0; kt < NKT; ++kt) {
    int buf = kt & 1;
#if USE_TDM
    if (w == 0) {
      if (kt + 1 < NKT) {   // DMA next tile into the other buffer
        tdm_load_2d_bf16((unsigned)(unsigned long long)(const void*)&kT[buf ^ 1][0],
                         Kh + (size_t)(kt + 1) * 32 * HDIM, 64, 32, 64, 32, 64);
        tdm_load_2d_bf16((unsigned)(unsigned long long)(const void*)&vT[buf ^ 1][0],
                         Vh + (kt + 1) * 32, 32, 64, 32, 64, SEQ);
        __builtin_amdgcn_s_wait_tensorcnt(2);   // tile kt's pair retired
      } else {
        __builtin_amdgcn_s_wait_tensorcnt(0);
      }
    }
    __syncthreads();        // tile kt visible to all waves
#else
    {
      int t = threadIdx.x;
      int r0 = t >> 3, c0 = (t & 7) * 8;      // 32x64 in 16B chunks
      *(u32x4*)&kT[buf][r0 * 64 + c0] =
          *(const u32x4*)(Kh + ((size_t)(kt * 32 + r0)) * 64 + c0);
      int r1 = t >> 2, c1 = (t & 3) * 8;      // 64x32 in 16B chunks
      *(u32x4*)&vT[buf][r1 * 32 + c1] =
          *(const u32x4*)(Vh + (size_t)r1 * SEQ + kt * 32 + c1);
    }
    __syncthreads();
#endif

    // S = Q * K^T for 16 queries x 32 keys (two 16x16 C tiles).
    v8f s0 = {}, s1 = {};
#pragma unroll
    for (int half = 0; half < 2; ++half) {
      AFrag bk0, bk1;
      const __bf16* kp = &kT[buf][(half * 16 + (lane & 15)) * 64 + kb16];
      bk0.q[0] = *(const u32x4*)kp;        bk0.q[1] = *(const u32x4*)(kp + 16);
      bk1.q[0] = *(const u32x4*)(kp + 32); bk1.q[1] = *(const u32x4*)(kp + 48);
      v8f c = {};
      c = wmma_bf16(aq0, bk0, c);
      c = wmma_bf16(aq1, bk1, c);
      if (half == 0) s0 = c; else s1 = c;
    }

    // Online softmax update (rows shared across each 16-lane half-wave).
    float sc[8];
#pragma unroll
    for (int r = 0; r < 8; ++r) {
      float t = fmaxf(s0[r], s1[r]);
      t = fmaxf(t, __shfl_xor(t, 1, 16));
      t = fmaxf(t, __shfl_xor(t, 2, 16));
      t = fmaxf(t, __shfl_xor(t, 4, 16));
      t = fmaxf(t, __shfl_xor(t, 8, 16));
      float mn = fmaxf(m[r], t);
      float rs = __expf(m[r] - mn);
      float p0 = __expf(s0[r] - mn);
      float p1 = __expf(s1[r] - mn);
      float u = p0 + p1;
      u += __shfl_xor(u, 1, 16);
      u += __shfl_xor(u, 2, 16);
      u += __shfl_xor(u, 4, 16);
      u += __shfl_xor(u, 8, 16);
      l[r] = l[r] * rs + u;
      m[r] = mn;
      sc[r] = rs;
      s0[r] = p0;
      s1[r] = p1;
    }
#pragma unroll
    for (int j = 0; j < 4; ++j)
#pragma unroll
      for (int r = 0; r < 8; ++r) o[j][r] *= sc[r];

    // Transpose P from C-layout to A-layout through per-wave LDS scratch.
    {
      int rr = 8 * (lane >> 4);
      int cc = lane & 15;
#pragma unroll
      for (int r = 0; r < 8; ++r) {
        pT[w][(rr + r) * 32 + cc]      = f2bf(s0[r]);
        pT[w][(rr + r) * 32 + 16 + cc] = f2bf(s1[r]);
      }
    }
    AFrag ap;
    {
      const __bf16* pp = &pT[w][(lane & 15) * 32 + kb8];
      ap.q[0] = *(const u32x4*)pp;
      ap.q[1] = *(const u32x4*)(pp + 16);
    }

    // O += P * V   (four 16x16 hd tiles, K-dim = 32 keys)
#pragma unroll
    for (int t4 = 0; t4 < 4; ++t4) {
      AFrag bv;
      const __bf16* vp = &vT[buf][(t4 * 16 + (lane & 15)) * 32 + kb16];
      bv.q[0] = *(const u32x4*)vp;
      bv.q[1] = *(const u32x4*)(vp + 16);
      o[t4] = wmma_bf16(ap, bv, o[t4]);
    }
    __syncthreads();   // everyone done with buf before it becomes a DMA target
  }

  // Normalize and store to Ob[b*N + n][h*64 + hd] (bf16, proj-GEMM layout).
  int rb = qblk * 128 + w * 16 + 8 * (lane >> 4);
  int bt = bh / NHEAD, h = bh % NHEAD;
#pragma unroll
  for (int r = 0; r < 8; ++r) {
    float inv = 1.0f / l[r];
    int n = rb + r;
    __bf16* op = Ob + ((size_t)bt * SEQ + n) * DMODEL + h * HDIM;
#pragma unroll
    for (int t4 = 0; t4 < 4; ++t4)
      op[t4 * 16 + (lane & 15)] = f2bf(o[t4][r] * inv);
  }
}

// ---------------------------------------------------------------------------
// Stage 3: projection GEMM out[8192,768] = Ob(bf16) @ w_proj(swizzled bf16)
//          + b_proj, f32 output straight to d_out. Register double-buffered.
// ---------------------------------------------------------------------------
__global__ __launch_bounds__(256) void proj_gemm_k(
    const __bf16* __restrict__ Ob, const __bf16* __restrict__ wsw,
    const float* __restrict__ bias, float* __restrict__ out) {
  const int NT = DMODEL / 16;               // 48
  int mblk = blockIdx.x;                    // 64
  int nblk = blockIdx.y;                    // 12
  int w    = threadIdx.x >> 5;
  int lane = threadIdx.x & 31;
  int kb8  = (lane >> 4) * 8;
  const __bf16* arow_p = Ob + (size_t)(mblk * 128 + w * 16 + (lane & 15)) * DMODEL + kb8;
  const __bf16* bbase  = wsw + ((size_t)(nblk * 4) * 32 + lane) * 16;

  auto loadA = [&](int ks, AFrag& a) {
    const __bf16* ap = arow_p + ks * 32;
    a.q[0] = *(const u32x4*)ap;
    a.q[1] = *(const u32x4*)(ap + 16);
  };
  auto loadB = [&](int ks, int j, AFrag& b) {
    const __bf16* bp = bbase + ((size_t)ks * NT + j) * 512;
    b.q[0] = *(const u32x4*)bp;
    b.q[1] = *(const u32x4*)(bp + 8);
  };

  v8f acc[4] = {};
  AFrag a0, a1, b0[4], b1[4];
  loadA(0, a0);
#pragma unroll
  for (int j = 0; j < 4; ++j) loadB(0, j, b0[j]);

  for (int ks = 0; ks < DMODEL / 32; ks += 2) {
    __builtin_prefetch(arow_p + (ks + 4) * 32, 0, 0);
    __builtin_prefetch(bbase + ((size_t)(ks + 4) * NT) * 512, 0, 0);
    loadA(ks + 1, a1);
#pragma unroll
    for (int j = 0; j < 4; ++j) loadB(ks + 1, j, b1[j]);
#pragma unroll
    for (int j = 0; j < 4; ++j) acc[j] = wmma_bf16(a0, b0[j], acc[j]);
    if (ks + 2 < DMODEL / 32) {
      loadA(ks + 2, a0);
#pragma unroll
      for (int j = 0; j < 4; ++j) loadB(ks + 2, j, b0[j]);
    }
#pragma unroll
    for (int j = 0; j < 4; ++j) acc[j] = wmma_bf16(a1, b1[j], acc[j]);
  }

  int rbase = mblk * 128 + w * 16 + 8 * (lane >> 4);
#pragma unroll
  for (int j = 0; j < 4; ++j) {
    int col = nblk * 64 + j * 16 + (lane & 15);
    float bv = bias[col];
#pragma unroll
    for (int r = 0; r < 8; ++r)
      out[(size_t)(rbase + r) * DMODEL + col] = acc[j][r] + bv;
  }
}

// ---------------------------------------------------------------------------
// Host launch
// ---------------------------------------------------------------------------
extern "C" void kernel_launch(void* const* d_in, const int* in_sizes, int n_in,
                              void* d_out, int out_size, void* d_ws, size_t ws_size,
                              hipStream_t stream) {
  const float* x      = (const float*)d_in[0];
  const float* w_qkv  = (const float*)d_in[1];
  const float* b_qkv  = (const float*)d_in[2];
  const float* w_proj = (const float*)d_in[3];
  const float* b_proj = (const float*)d_in[4];
  float* out = (float*)d_out;

  char* ws = (char*)d_ws;
  size_t off = 0;
  auto alloc = [&](size_t bytes) -> void* {
    void* p = ws + off;
    off += (bytes + 255) & ~(size_t)255;
    return p;
  };

  const size_t nX = (size_t)ROWS * DMODEL;                 // 6,291,456
  const size_t nWqkv = (size_t)DMODEL * 3 * DMODEL;        // 1,769,472
  const size_t nWproj = (size_t)DMODEL * DMODEL;           //   589,824
  const size_t nQKV = (size_t)BATCH * NHEAD * SEQ * HDIM;  // 6,291,456 each

  __bf16* xb   = (__bf16*)alloc(nX * 2);
  __bf16* wqs  = (__bf16*)alloc(nWqkv * 2);
  __bf16* wps  = (__bf16*)alloc(nWproj * 2);
  __bf16* Qb   = (__bf16*)alloc(nQKV * 2);
  __bf16* Kb   = (__bf16*)alloc(nQKV * 2);
  __bf16* Vtb  = (__bf16*)alloc(nQKV * 2);
  __bf16* Ob   = (__bf16*)alloc(nX * 2);

  cvt_f32_bf16_k<<<(unsigned)((nX + 255) / 256), 256, 0, stream>>>(x, xb, (int)nX);
  swizzle_w_k<<<(unsigned)((nWqkv + 255) / 256), 256, 0, stream>>>(w_qkv, wqs, DMODEL, 3 * DMODEL);
  swizzle_w_k<<<(unsigned)((nWproj + 255) / 256), 256, 0, stream>>>(w_proj, wps, DMODEL, DMODEL);

  qkv_gemm_k<<<dim3(ROWS / 128, (3 * DMODEL) / 64), 256, 0, stream>>>(xb, wqs, b_qkv, Qb, Kb, Vtb);
  attn_k<<<dim3(BATCH * NHEAD, SEQ / 128), 256, 0, stream>>>(Qb, Kb, Vtb, Ob);
  proj_gemm_k<<<dim3(ROWS / 128, DMODEL / 64), 256, 0, stream>>>(Ob, wps, b_proj, out);
}
